// GIN_17128329576567
// MI455X (gfx1250) — matrix operand order repaired and checked by reading, not compiled, not run on hardware
//
#include <hip/hip_runtime.h>
#include <hip/hip_bf16.h>

#define N_NODES_C 100000
#define N_EDGES_C 1600000
#define FEATS_C   128
#define MTILES_C  (N_NODES_C / 16)   // 6250, exact

typedef __attribute__((ext_vector_type(16))) _Float16 v16h;
typedef __attribute__((ext_vector_type(8)))  float    v8f;

// ---- order-preserving float<->uint encoding for atomic max over signed floats
__device__ __forceinline__ unsigned enc_f32(float f) {
  unsigned u = __float_as_uint(f);
  return (u & 0x80000000u) ? ~u : (u | 0x80000000u);
}
__device__ __forceinline__ float dec_f32(unsigned u) {
  return __uint_as_float((u & 0x80000000u) ? (u ^ 0x80000000u) : ~u);
}

// ---- in-degree via u32 atomics
__global__ void gin_deg_kernel(const int* __restrict__ dst, unsigned* __restrict__ deg) {
  int e = blockIdx.x * blockDim.x + threadIdx.x;
  if (e < N_EDGES_C) atomicAdd(&deg[dst[e]], 1u);
}

// ---- edge-parallel scatter-add: one wave lane handles 4 contiguous feats
__global__ void gin_agg_sum_kernel(const float* __restrict__ h, const int* __restrict__ src,
                                   const int* __restrict__ dst, float* __restrict__ agg) {
  int idx = blockIdx.x * blockDim.x + threadIdx.x;   // E*32 threads
  int e = idx >> 5;
  int lane = idx & 31;
  if (e >= N_EDGES_C) return;
  int s = src[e], d = dst[e];
  const float4 v = *(const float4*)(h + (long long)s * FEATS_C + lane * 4);
  float* p = agg + (long long)d * FEATS_C + lane * 4;
  atomicAdd(p + 0, v.x);
  atomicAdd(p + 1, v.y);
  atomicAdd(p + 2, v.z);
  atomicAdd(p + 3, v.w);
}

// ---- edge-parallel scatter-max via monotone uint encoding (init 0 == -inf)
__global__ void gin_agg_max_kernel(const float* __restrict__ h, const int* __restrict__ src,
                                   const int* __restrict__ dst, unsigned* __restrict__ agg) {
  int idx = blockIdx.x * blockDim.x + threadIdx.x;
  int e = idx >> 5;
  int lane = idx & 31;
  if (e >= N_EDGES_C) return;
  int s = src[e], d = dst[e];
  const float4 v = *(const float4*)(h + (long long)s * FEATS_C + lane * 4);
  unsigned* p = agg + (long long)d * FEATS_C + lane * 4;
  atomicMax(p + 0, enc_f32(v.x));
  atomicMax(p + 1, enc_f32(v.y));
  atomicMax(p + 2, enc_f32(v.z));
  atomicMax(p + 3, enc_f32(v.w));
}

// ---- combine (1+eps)*h + agg, fix up max/mean, emit packed f16 pairs straight
//      into the CDNA5 16-bit A-matrix 16x32 fragment layout (CDNA5 ISA 7.12.2).
//      One thread per (node, even-K pair) -> single dword store.
//      Also resets agg to 0 so the next layer's atomic pass starts clean.
//      mode: 0=max, 1=sum, 2=mean
__global__ void gin_combine_kernel(const float* __restrict__ h, unsigned* __restrict__ aggu,
                                   const unsigned* __restrict__ deg, const float* __restrict__ eps,
                                   int epsIdx, int mode, unsigned* __restrict__ Apack) {
  int idx = blockIdx.x * blockDim.x + threadIdx.x;   // N_NODES*64 threads
  if (idx >= N_NODES_C * (FEATS_C / 2)) return;
  int kp   = idx & 63;          // K-pair index
  int node = idx >> 6;
  int k = kp * 2;               // even K
  long long base = (long long)node * FEATS_C + k;
  unsigned dg = deg[node];
  uint2 au = *(const uint2*)(aggu + base);
  *(uint2*)(aggu + base) = make_uint2(0u, 0u);       // pre-zero for next layer
  float a0, a1;
  if (mode == 0) {
    a0 = (dg > 0u) ? dec_f32(au.x) : 0.0f;
    a1 = (dg > 0u) ? dec_f32(au.y) : 0.0f;
  } else {
    float s0 = __uint_as_float(au.x);
    float s1 = __uint_as_float(au.y);
    if (mode == 1) { a0 = s0; a1 = s1; }
    else {
      float inv = 1.0f / (float)(dg > 1u ? dg : 1u);
      a0 = s0 * inv; a1 = s1 * inv;
    }
  }
  float ep = 1.0f + eps[epsIdx];
  float2 hv = *(const float2*)(h + base);
  _Float16 f0 = (_Float16)(ep * hv.x + a0);
  _Float16 f1 = (_Float16)(ep * hv.y + a1);
  unsigned u0 = (unsigned)__builtin_bit_cast(unsigned short, f0);
  unsigned u1 = (unsigned)__builtin_bit_cast(unsigned short, f1);
  // fragment scatter: lanes 0-15 hold K 0-7 (v0-3) & 16-23 (v4-7); lanes 16-31 hold K 8-15 & 24-31
  int tile = node >> 4, lane_lo = node & 15;
  int ks = k >> 5, kk = k & 31;
  int group = kk >> 3;
  int lane = lane_lo + ((group & 1) << 4);
  int v = ((group >> 1) << 2) + ((kk & 7) >> 1);
  Apack[((tile * 4 + ks) * 32 + lane) * 8 + v] = (u1 << 16) | u0;  // low half = even K
}

// ---- pack a row-major f32 weight [128 x Ncols] into B-matrix 32x16 f16 fragments:
//      lane<16: col=lane, v<4 -> K=2v,2v+1, v>=4 -> K=16+2(v-4),+1 ; lane>=16: col=lane-16, same K +8 (per K-step of 32)
__global__ void gin_pack_w_kernel(const float* __restrict__ W, unsigned* __restrict__ Bpack,
                                  int Ncols, int Ntiles) {
  int idx = blockIdx.x * blockDim.x + threadIdx.x;   // Ntiles*4*32*8 threads
  if (idx >= Ntiles * 1024) return;
  int v = idx & 7, lane = (idx >> 3) & 31, ks = (idx >> 8) & 3, nt = idx >> 10;
  int col = nt * 16 + (lane & 15);
  int k = ks * 32 + ((lane >> 4) << 3) + ((v >> 2) << 4) + 2 * (v & 3);
  _Float16 h0 = (_Float16)W[k * Ncols + col];
  _Float16 h1 = (_Float16)W[(k + 1) * Ncols + col];
  unsigned u0 = (unsigned)__builtin_bit_cast(unsigned short, h0);
  unsigned u1 = (unsigned)__builtin_bit_cast(unsigned short, h1);
  Bpack[idx] = (u1 << 16) | u0;   // low half = even K
}

// ---- WMMA GEMM: each wave owns 16 rows; 4 K-steps x Ntiles N-tiles of
//      v_wmma_f32_16x16x32_f16; bias fused into accumulator init; optional ReLU
__global__ __launch_bounds__(256) void gin_gemm_wmma_kernel(
    const v16h* __restrict__ Apack, const v16h* __restrict__ Bpack,
    const float* __restrict__ bias, float* __restrict__ out,
    int Ntiles, int Ncols, int doRelu) {
  int wave = (blockIdx.x * blockDim.x + threadIdx.x) >> 5;
  int lane = threadIdx.x & 31;
  if (wave >= MTILES_C) return;        // whole-wave exit: EXEC stays all-ones for WMMA
  v16h a0 = Apack[(wave * 4 + 0) * 32 + lane];
  v16h a1 = Apack[(wave * 4 + 1) * 32 + lane];
  v16h a2 = Apack[(wave * 4 + 2) * 32 + lane];
  v16h a3 = Apack[(wave * 4 + 3) * 32 + lane];
  int colIn = lane & 15;
  int mhalf = lane >> 4;
  for (int nt = 0; nt < Ntiles; ++nt) {
    v16h b0 = Bpack[(nt * 4 + 0) * 32 + lane];
    v16h b1 = Bpack[(nt * 4 + 1) * 32 + lane];
    v16h b2 = Bpack[(nt * 4 + 2) * 32 + lane];
    v16h b3 = Bpack[(nt * 4 + 3) * 32 + lane];
    float bv = bias[nt * 16 + colIn];
    v8f acc = {bv, bv, bv, bv, bv, bv, bv, bv};
    acc = __builtin_amdgcn_wmma_f32_16x16x32_f16(false, a0, false, b0, (short)0, acc, false, false);
    acc = __builtin_amdgcn_wmma_f32_16x16x32_f16(false, a1, false, b1, (short)0, acc, false, false);
    acc = __builtin_amdgcn_wmma_f32_16x16x32_f16(false, a2, false, b2, (short)0, acc, false, false);
    acc = __builtin_amdgcn_wmma_f32_16x16x32_f16(false, a3, false, b3, (short)0, acc, false, false);
    int rowBase = wave * 16 + mhalf * 8;   // C/D layout: VGPR r -> M = 8*(lane/16)+r
    int colBase = nt * 16 + colIn;
#pragma unroll
    for (int r = 0; r < 8; ++r) {
      float x = acc[r];
      if (doRelu) x = x > 0.0f ? x : 0.0f;
      out[(long long)(rowBase + r) * Ncols + colBase] = x;
    }
  }
}

extern "C" void kernel_launch(void* const* d_in, const int* in_sizes, int n_in,
                              void* d_out, int out_size, void* d_ws, size_t ws_size,
                              hipStream_t stream) {
  const float* features = (const float*)d_in[0];
  const int*   src      = (const int*)d_in[1];
  const int*   dst      = (const int*)d_in[2];
  const float* W1  = (const float*)d_in[3];  const float* b1  = (const float*)d_in[4];
  const float* W2a = (const float*)d_in[5];  const float* b2a = (const float*)d_in[6];
  const float* W2b = (const float*)d_in[7];  const float* b2b = (const float*)d_in[8];
  const float* W3  = (const float*)d_in[9];  const float* b3  = (const float*)d_in[10];
  const float* eps = (const float*)d_in[11];

  // workspace layout (all offsets 256B-aligned); ~129 MB total
  char* ws = (char*)d_ws;
  const size_t HBYTES = (size_t)N_NODES_C * FEATS_C * 4;     // 51,200,000
  unsigned* deg   = (unsigned*)(ws + 0);                     // 400 KB region
  float*    h     = (float*)(ws + 524288);
  float*    agg   = (float*)(ws + 524288 + HBYTES);
  unsigned* Apack = (unsigned*)(ws + 524288 + 2 * HBYTES);   // 25.6 MB
  unsigned* Bp1   = (unsigned*)(ws + 524288 + 2 * HBYTES + 25600000);
  unsigned* Bp2a  = Bp1 + 8192;
  unsigned* Bp2b  = Bp2a + 8192;
  unsigned* Bp3   = Bp2b + 8192;

  const int edgeGrid = (N_EDGES_C * 32) / 256;             // 200000
  const int combGrid = (N_NODES_C * (FEATS_C / 2)) / 256;  // 25000
  const int gemmGrid = (MTILES_C + 7) / 8;                 // 8 waves / block

  hipMemsetAsync(deg, 0, (size_t)N_NODES_C * 4, stream);
  hipMemsetAsync(agg, 0, HBYTES, stream);            // 0 == encoded -inf for max

  gin_deg_kernel<<<(N_EDGES_C + 255) / 256, 256, 0, stream>>>(dst, deg);
  gin_pack_w_kernel<<<32, 256, 0, stream>>>(W1,  Bp1,  128, 8);
  gin_pack_w_kernel<<<32, 256, 0, stream>>>(W2a, Bp2a, 128, 8);
  gin_pack_w_kernel<<<32, 256, 0, stream>>>(W2b, Bp2b, 128, 8);
  gin_pack_w_kernel<<<16, 256, 0, stream>>>(W3,  Bp3,  64,  4);

  // layer 0: max aggregation -> Linear(W1)+ReLU   (combine re-zeros agg)
  gin_agg_max_kernel<<<edgeGrid, 256, 0, stream>>>(features, src, dst, (unsigned*)agg);
  gin_combine_kernel<<<combGrid, 256, 0, stream>>>(features, (unsigned*)agg, deg, eps, 0, 0, Apack);
  gin_gemm_wmma_kernel<<<gemmGrid, 256, 0, stream>>>((const v16h*)Apack, (const v16h*)Bp1, b1, h, 8, 128, 1);

  // layer 1: sum aggregation -> Linear(W2a)+ReLU
  gin_agg_sum_kernel<<<edgeGrid, 256, 0, stream>>>(h, src, dst, agg);
  gin_combine_kernel<<<combGrid, 256, 0, stream>>>(h, (unsigned*)agg, deg, eps, 1, 1, Apack);
  gin_gemm_wmma_kernel<<<gemmGrid, 256, 0, stream>>>((const v16h*)Apack, (const v16h*)Bp2a, b2a, h, 8, 128, 1);

  // layer 2: sum aggregation -> Linear(W2b)+ReLU
  gin_agg_sum_kernel<<<edgeGrid, 256, 0, stream>>>(h, src, dst, agg);
  gin_combine_kernel<<<combGrid, 256, 0, stream>>>(h, (unsigned*)agg, deg, eps, 2, 1, Apack);
  gin_gemm_wmma_kernel<<<gemmGrid, 256, 0, stream>>>((const v16h*)Apack, (const v16h*)Bp2b, b2b, h, 8, 128, 1);

  // output layer: mean aggregation -> Linear(W3), no activation, f32 out
  gin_agg_sum_kernel<<<edgeGrid, 256, 0, stream>>>(h, src, dst, agg);
  gin_combine_kernel<<<combGrid, 256, 0, stream>>>(h, (unsigned*)agg, deg, eps, 3, 2, Apack);
  gin_gemm_wmma_kernel<<<gemmGrid, 256, 0, stream>>>((const v16h*)Apack, (const v16h*)Bp3, b3, (float*)d_out, 4, 64, 0);
}
